// ExponentialGlobalConv_40123584479318
// MI455X (gfx1250) — matrix operand order repaired and checked by reading, not compiled, not run on hardware
//
#include <hip/hip_runtime.h>
#include <hip/hip_bf16.h>

#define NUM_CHANNELS 32
#define NUM_GRIDS    2048
#define BATCH        32
#define MIN_XI       0.1f
#define MAX_XI       5.0f
#define DX           0.08f

#define TAB_LEN      4096   // extended symmetric Toeplitz row: i -> |i-2047|
#define TAB_CENTER   2047

typedef __attribute__((ext_vector_type(16))) __bf16 v16bf;
typedef __attribute__((ext_vector_type(8)))  __bf16 v8bf;
typedef __bf16 v8bf_u __attribute__((ext_vector_type(8), aligned(2)));  // unaligned LDS view
typedef __attribute__((ext_vector_type(8)))  float  v8f;
typedef __attribute__((ext_vector_type(4)))  int    v4i;

#define AS_GLOBAL __attribute__((address_space(1)))
#define AS_LDS    __attribute__((address_space(3)))

#if __has_builtin(__builtin_amdgcn_global_load_async_to_lds_b128) && \
    __has_builtin(__builtin_amdgcn_s_wait_asynccnt)
#define EGC_ASYNC 1
#else
#define EGC_ASYNC 0
#endif

// widths[c] = MIN + (MAX-MIN)*sigmoid(eta[c])
__device__ __forceinline__ float width_of(const float* eta, int c) {
    float e = eta[c];
    float s = 1.0f / (1.0f + __expf(-e));
    return MIN_XI + (MAX_XI - MIN_XI) * s;
}

// Convert inputs to bf16 and build extended symmetric Toeplitz rows:
//   tabExt[c][i] = exp(-|i-2047|*DX/w_c) * DX/(2*w_c)
__global__ void egc_prep_kernel(const float* __restrict__ inputs,
                                const float* __restrict__ eta,
                                __bf16* __restrict__ inBf,
                                __bf16* __restrict__ tabExt) {
    int idx = blockIdx.x * blockDim.x + threadIdx.x;
    if (idx < BATCH * NUM_GRIDS) {
        inBf[idx] = (__bf16)inputs[idx];
    }
    if (idx < NUM_CHANNELS * TAB_LEN) {
        int c = idx >> 12;               // TAB_LEN == 4096
        int i = idx & (TAB_LEN - 1);
        int d = i - TAB_CENTER;
        d = d < 0 ? -d : d;
        float w = width_of(eta, c);
        float v = __expf(-(float)d * (DX / w)) * (DX / (2.0f * w));
        tabExt[idx] = (__bf16)v;
    }
}

// One wave per (16-wide m tile, pair of channels).
// D = A(32x2048 bf16) x Toeplitz(2048x16) per channel via v_wmma_f32_16x16x32_bf16.
// 4 WMMAs per k-step (2 batch halves x 2 channels) share the same A loads.
__global__ void __launch_bounds__(32)
egc_wmma_kernel(const __bf16* __restrict__ inBf,
                const __bf16* __restrict__ tabExt,
                float* __restrict__ out) {
    __shared__ __bf16 smem0[TAB_LEN];   // 8 KB: channel c0 extended row
    __shared__ __bf16 smem1[TAB_LEN];   // 8 KB: channel c0+1 extended row

    const int c0   = blockIdx.y * 2;
    const int m0   = blockIdx.x * 16;
    const int lane = threadIdx.x;        // 0..31 (wave32)
    const int half = lane >> 4;
    const int l16  = lane & 15;

    const __bf16* row0 = tabExt + (size_t)c0 * TAB_LEN;
    const __bf16* row1 = row0 + TAB_LEN;

#if EGC_ASYNC
    // CDNA5 async copy: global -> LDS, tracked by ASYNCcnt (no VGPR round trip).
    for (int i = lane * 8; i < TAB_LEN; i += 32 * 8) {
        __builtin_amdgcn_global_load_async_to_lds_b128(
            (AS_GLOBAL v4i*)(v4i*)(row0 + i),
            (AS_LDS    v4i*)(v4i*)(&smem0[i]), 0, 0);
        __builtin_amdgcn_global_load_async_to_lds_b128(
            (AS_GLOBAL v4i*)(v4i*)(row1 + i),
            (AS_LDS    v4i*)(v4i*)(&smem1[i]), 0, 0);
    }
    __builtin_amdgcn_s_wait_asynccnt(0);
#else
    for (int i = lane * 8; i < TAB_LEN; i += 32 * 8) {
        *(v8bf*)(&smem0[i]) = *(const v8bf*)(&row0[i]);
        *(v8bf*)(&smem1[i]) = *(const v8bf*)(&row1[i]);
    }
#endif
    __syncthreads();

    v8f accA0 = {};   // batches 0..15,  channel c0
    v8f accA1 = {};   // batches 0..15,  channel c0+1
    v8f accB0 = {};   // batches 16..31, channel c0
    v8f accB1 = {};   // batches 16..31, channel c0+1
    const int m = m0 + l16;

    for (int k0 = 0; k0 < NUM_GRIDS; k0 += 32) {
        // A layout (ISA 7.12.2, 16-bit A 16x32): row M = lane%16,
        // element e holds K = k0 + 8*half + (e<8 ? e : e+8) -> two aligned 16B runs.
        const __bf16* a0p = inBf + (size_t)l16        * NUM_GRIDS + k0 + half * 8;
        const __bf16* a1p = inBf + (size_t)(l16 + 16) * NUM_GRIDS + k0 + half * 8;
        v8bf a0lo = *(const v8bf*)(a0p);
        v8bf a0hi = *(const v8bf*)(a0p + 16);
        v8bf a1lo = *(const v8bf*)(a1p);
        v8bf a1hi = *(const v8bf*)(a1p + 16);
        v16bf A0 = __builtin_shufflevector(a0lo, a0hi, 0,1,2,3,4,5,6,7,8,9,10,11,12,13,14,15);
        v16bf A1 = __builtin_shufflevector(a1lo, a1hi, 0,1,2,3,4,5,6,7,8,9,10,11,12,13,14,15);

        // B layout (32x16, K x N): column N = lane%16, element e holds
        // K = k0 + 8*half + (e<8 ? e : e+8). Extended table makes each 8-run contiguous:
        //   smem[(kb + e) - m + TAB_CENTER].
        const int idx0 = k0 + 8 * half - m + TAB_CENTER;
        v8bf_u b0lo = *(const v8bf_u*)(&smem0[idx0]);
        v8bf_u b0hi = *(const v8bf_u*)(&smem0[idx0 + 16]);
        v8bf_u b1lo = *(const v8bf_u*)(&smem1[idx0]);
        v8bf_u b1hi = *(const v8bf_u*)(&smem1[idx0 + 16]);
        v16bf B0 = __builtin_shufflevector(b0lo, b0hi, 0,1,2,3,4,5,6,7,8,9,10,11,12,13,14,15);
        v16bf B1 = __builtin_shufflevector(b1lo, b1hi, 0,1,2,3,4,5,6,7,8,9,10,11,12,13,14,15);

        accA0 = __builtin_amdgcn_wmma_f32_16x16x32_bf16(false, A0, false, B0,
                                                        (short)0, accA0, false, false);
        accA1 = __builtin_amdgcn_wmma_f32_16x16x32_bf16(false, A0, false, B1,
                                                        (short)0, accA1, false, false);
        accB0 = __builtin_amdgcn_wmma_f32_16x16x32_bf16(false, A1, false, B0,
                                                        (short)0, accB0, false, false);
        accB1 = __builtin_amdgcn_wmma_f32_16x16x32_bf16(false, A1, false, B1,
                                                        (short)0, accB1, false, false);
    }

    // D layout: VGPR r, lane L -> M = r + 8*(L>=16) (= batch), N = L%16 (= m).
    #pragma unroll
    for (int r = 0; r < 8; ++r) {
        int b0 = r + 8 * half;
        size_t o0 = ((size_t)b0        * NUM_GRIDS + m) * NUM_CHANNELS;
        size_t o1 = ((size_t)(b0 + 16) * NUM_GRIDS + m) * NUM_CHANNELS;
        out[o0 + c0]     = accA0[r];
        out[o0 + c0 + 1] = accA1[r];
        out[o1 + c0]     = accB0[r];
        out[o1 + c0 + 1] = accB1[r];
    }
}

extern "C" void kernel_launch(void* const* d_in, const int* in_sizes, int n_in,
                              void* d_out, int out_size, void* d_ws, size_t ws_size,
                              hipStream_t stream) {
    const float* inputs = (const float*)d_in[0];   // (B, N) fp32
    // d_in[1] = grids: uniform spacing DX by construction -> Toeplitz, unused
    const float* eta    = (const float*)d_in[2];   // (C,) fp32

    __bf16* inBf   = (__bf16*)d_ws;                       // B*N bf16        (128 KB)
    __bf16* tabExt = inBf + (size_t)BATCH * NUM_GRIDS;    // C*TAB_LEN bf16  (256 KB)

    const int prep_total = NUM_CHANNELS * TAB_LEN;        // 131072 (covers inputs too)
    egc_prep_kernel<<<(prep_total + 255) / 256, 256, 0, stream>>>(inputs, eta, inBf, tabExt);

    dim3 grid(NUM_GRIDS / 16, NUM_CHANNELS / 2);   // 128 m-tiles x 16 channel pairs
    egc_wmma_kernel<<<grid, 32, 0, stream>>>(inBf, tabExt, (float*)d_out);
}
